// LinAngularAttention_1640677507597
// MI455X (gfx1250) — compile-verified, matrix operand order checked by reading, not run on hardware
//
#include <hip/hip_runtime.h>
#include <hip/hip_bf16.h>

// ---------------------------------------------------------------------------
// LinAngularAttention for MI455X (gfx1250, wave32, WMMA + TDM)
// N=4, L=4096, C=512, H=8, D=64, KS=9
// ---------------------------------------------------------------------------

#define NB_     4
#define L_      4096
#define C_      512
#define H_      8
#define D_      64
#define KS_     9
#define INV_PI_ 0.31830988618379067f

typedef __attribute__((ext_vector_type(16))) _Float16 v16h;
typedef __attribute__((ext_vector_type(8)))  float    v8f;

union FragU {
    v16h         v;
    unsigned int u[8];
    _Float16     h[16];
};

union Pack2 {
    _Float16     h[2];
    unsigned int u;
};

// K index of the low element of packed pair j (0..7) for this lane half.
// ISA 16-bit A 16x32 layout: lanes0-15 VGPR0..3 -> K=0,1 / 2,3 / 4,5 / 6,7,
// VGPR4..7 -> K=16..23; lanes16-31 get the +8 K-half.  B mirrored (N in lanes).
__device__ __forceinline__ int frag_k0(int j, int khalf) {
    return ((j & 4) << 2) + (khalf << 3) + ((j & 3) << 1);
}

// A-style fragment: per-lane row (row+lane&15), K contiguous in memory.
__device__ __forceinline__ v16h load_frag_rowK(const _Float16* base, int row,
                                               int stride, int kbase, int lane) {
    FragU f;
    int mrow  = lane & 15;
    int khalf = (lane >> 4) & 1;
    const _Float16* p = base + (size_t)(row + mrow) * stride + kbase;
#pragma unroll
    for (int j = 0; j < 8; ++j) {
        int k0 = frag_k0(j, khalf);
        f.u[j] = *(const unsigned int*)(p + k0);
    }
    return f.v;
}

// B-style fragment (also A-from-transposed-source): source row-major [K][stride],
// lane column = col0 + (lane&15), K runs across rows -> strided 2-byte loads.
__device__ __forceinline__ v16h load_frag_colK(const _Float16* base, int kbase,
                                               int stride, int col0, int lane) {
    FragU f;
    int n     = col0 + (lane & 15);
    int khalf = (lane >> 4) & 1;
#pragma unroll
    for (int j = 0; j < 8; ++j) {
        int k0 = frag_k0(j, khalf);
        f.h[2 * j]     = base[(size_t)(kbase + k0) * stride + n];
        f.h[2 * j + 1] = base[(size_t)(kbase + k0 + 1) * stride + n];
    }
    return f.v;
}

__device__ __forceinline__ v8f wmma_f16(v16h a, v16h b, v8f c) {
    return __builtin_amdgcn_wmma_f32_16x16x32_f16(
        false, a, false, b, (short)0, c, false, false);
}

// ---------------------------------------------------------------------------
// Tensor Data Mover: 2D tile (rows x row_units8 * 8B) global -> LDS.
// Descriptor per CDNA5 ISA (D# group0/group1).
// ---------------------------------------------------------------------------
#if __has_builtin(__builtin_amdgcn_tensor_load_to_lds) && \
    __has_builtin(__builtin_amdgcn_s_wait_tensorcnt)
#define USE_TDM 1
typedef unsigned int u32x4 __attribute__((ext_vector_type(4)));
typedef int          i32x4 __attribute__((ext_vector_type(4)));
typedef int          i32x8 __attribute__((ext_vector_type(8)));

__device__ __forceinline__ void tdm_load_2d(unsigned lds_addr, const void* gaddr,
                                            unsigned row_units8, unsigned rows) {
    unsigned long long ga = (unsigned long long)(size_t)gaddr;
    u32x4 g0;
    g0[0] = 1u;                                             // count=1 (valid user D#)
    g0[1] = lds_addr;                                       // [63:32] lds byte addr
    g0[2] = (unsigned)(ga & 0xffffffffu);                   // [95:64] global addr lo
    g0[3] = (unsigned)((ga >> 32) & 0x01ffffffu)            // [120:96] global addr hi
          | 0x80000000u;                                    // [127:126] type=2
    i32x8 g1;
    g1[0] = (int)(3u << 16);                                // data_size=3 (8B), no flags
    g1[1] = (int)(row_units8 << 16);                        // tensor_dim0[15:0]
    g1[2] = (int)(rows << 16);                              // tensor_dim1[15:0]
    g1[3] = (int)(row_units8 << 16);                        // tile_dim0
    g1[4] = (int)rows;                                      // tile_dim1 (tile_dim2=0)
    g1[5] = (int)row_units8;                                // tensor_dim0_stride lo32
    g1[6] = 0;
    g1[7] = 0;
    i32x4 z4 = {0, 0, 0, 0};
#if __clang_major__ >= 23
    i32x8 z8 = {0, 0, 0, 0, 0, 0, 0, 0};
    __builtin_amdgcn_tensor_load_to_lds(g0, g1, z4, z4, z8, 0);
#else
    __builtin_amdgcn_tensor_load_to_lds(g0, g1, z4, z4, 0);
#endif
}
#else
#define USE_TDM 0
#endif

// ---------------------------------------------------------------------------
// Kernel 1: qkv = x @ w_qkv + b_qkv, scattered into q/k/v [N,H,L,D] f32.
// Tile: 128(M) x 64(N) x 32(K), 8 waves, each wave a 32x32 subtile (2x2 wmma).
// ---------------------------------------------------------------------------
__global__ __launch_bounds__(256)
void qkv_gemm_kernel(const float* __restrict__ x, const float* __restrict__ w,
                     const float* __restrict__ bias,
                     float* __restrict__ q32, float* __restrict__ k32,
                     float* __restrict__ v32) {
    __shared__ __align__(16) _Float16 As[128 * 32];   // [M][K]
    __shared__ __align__(16) _Float16 Bs[64 * 32];    // transposed: [N][K]

    int tid  = threadIdx.x;
    int lane = tid & 31, wave = tid >> 5;
    int bm = blockIdx.x * 128;
    int bn = blockIdx.y * 64;
    int wm = (wave & 3) * 32;
    int wn = (wave >> 2) * 32;

    v8f acc[2][2] = {};

    for (int kb = 0; kb < C_; kb += 32) {
        // Stage A: 128x32 f32 -> f16, packed pairs (float2 load, b32 LDS store).
#pragma unroll
        for (int i = 0; i < 8; ++i) {
            int p = tid + i * 256;                     // 2048 pairs
            int r = p >> 4, c2 = (p & 15) << 1;
            const float2 xv = *(const float2*)(x + (size_t)(bm + r) * C_ + kb + c2);
            Pack2 pk; pk.h[0] = (_Float16)xv.x; pk.h[1] = (_Float16)xv.y;
            ((unsigned int*)As)[p] = pk.u;
        }
        // Stage B transposed: Bs[n][k] = w[kb+k][bn+n], packed K pairs.
#pragma unroll
        for (int i = 0; i < 4; ++i) {
            int p = tid + i * 256;                     // 1024 pairs
            int n = p & 63, c2 = (p >> 6) << 1;
            float b0 = w[(size_t)(kb + c2) * (3 * C_) + bn + n];
            float b1 = w[(size_t)(kb + c2 + 1) * (3 * C_) + bn + n];
            Pack2 pk; pk.h[0] = (_Float16)b0; pk.h[1] = (_Float16)b1;
            ((unsigned int*)Bs)[n * 16 + (c2 >> 1)] = pk.u;
        }
        __syncthreads();

        v16h af[2], bf[2];
#pragma unroll
        for (int i = 0; i < 2; ++i) af[i] = load_frag_rowK(As, wm + i * 16, 32, 0, lane);
#pragma unroll
        for (int j = 0; j < 2; ++j) bf[j] = load_frag_rowK(Bs, wn + j * 16, 32, 0, lane);
#pragma unroll
        for (int i = 0; i < 2; ++i)
#pragma unroll
            for (int j = 0; j < 2; ++j)
                acc[i][j] = wmma_f16(af[i], bf[j], acc[i][j]);
        __syncthreads();
    }

    // Epilogue: bias + scatter into q/k/v [N,H,L,D].
    int mhi  = (lane >> 4) ? 8 : 0;
    int ncol = lane & 15;
#pragma unroll
    for (int i = 0; i < 2; ++i) {
#pragma unroll
        for (int j = 0; j < 2; ++j) {
#pragma unroll
            for (int r = 0; r < 8; ++r) {
                int m = bm + wm + i * 16 + mhi + r;
                int c = bn + wn + j * 16 + ncol;
                float val = acc[i][j][r] + bias[c];
                int nI = m >> 12, l = m & (L_ - 1);
                int which = c >> 9, rem = c & (C_ - 1);
                int h = rem >> 6, d = rem & (D_ - 1);
                size_t off = ((size_t)(nI * H_ + h) * L_ + l) * D_ + d;
                float* dst = (which == 0) ? q32 : ((which == 1) ? k32 : v32);
                dst[off] = val;
            }
        }
    }
}

// ---------------------------------------------------------------------------
// Kernel 2: L2-normalize q,k along D (64); emit f16 q,k (normalized) and v.
// ---------------------------------------------------------------------------
__global__ __launch_bounds__(256)
void norm_cast_kernel(const float* __restrict__ q32, const float* __restrict__ k32,
                      const float* __restrict__ v32,
                      _Float16* __restrict__ qh, _Float16* __restrict__ kh,
                      _Float16* __restrict__ vh) {
    int tid = threadIdx.x, lane = tid & 31, wave = tid >> 5;
    size_t row  = (size_t)blockIdx.x * 8 + wave;     // N*H*L rows
    size_t base = row * D_;

    float q0 = q32[base + lane], q1 = q32[base + lane + 32];
    float k0 = k32[base + lane], k1 = k32[base + lane + 32];
    float v0 = v32[base + lane], v1 = v32[base + lane + 32];

    float sq = q0 * q0 + q1 * q1;
    float sk = k0 * k0 + k1 * k1;
#pragma unroll
    for (int m = 16; m >= 1; m >>= 1) {
        sq += __shfl_xor(sq, m, 32);
        sk += __shfl_xor(sk, m, 32);
    }
    float iq = 1.0f / sqrtf(sq);
    float ik = 1.0f / sqrtf(sk);

    qh[base + lane]      = (_Float16)(q0 * iq);
    qh[base + lane + 32] = (_Float16)(q1 * iq);
    kh[base + lane]      = (_Float16)(k0 * ik);
    kh[base + lane + 32] = (_Float16)(k1 * ik);
    vh[base + lane]      = (_Float16)v0;
    vh[base + lane + 32] = (_Float16)v1;
}

// ---------------------------------------------------------------------------
// Kernel 3: attn[n,h] = K^T V  (64x64, reduce over L=4096).
// One block per (n,h); 64-row chunks of k/v staged in LDS (TDM when available);
// each wave owns 2 of the 16 output 16x16 tiles.
// ---------------------------------------------------------------------------
__global__ __launch_bounds__(256)
void attn_kernel(const _Float16* __restrict__ kh, const _Float16* __restrict__ vh,
                 float* __restrict__ attn) {
    __shared__ __align__(16) _Float16 Ks[64 * 64];
    __shared__ __align__(16) _Float16 Vs[64 * 64];

    int nh  = blockIdx.x;
    int tid = threadIdx.x, lane = tid & 31, wave = tid >> 5;
    int tm  = (wave & 3) * 16;
    int tn0 = (wave >> 2) * 32;

    const _Float16* kp = kh + (size_t)nh * L_ * D_;
    const _Float16* vp = vh + (size_t)nh * L_ * D_;

#if USE_TDM
    unsigned ldsK = (unsigned)(size_t)&Ks[0];
    unsigned ldsV = (unsigned)(size_t)&Vs[0];
#endif

    v8f acc[2] = {};

    for (int l0 = 0; l0 < L_; l0 += 64) {
#if USE_TDM
        // Async DMA: two 64-row x 128B tiles, issued once per workgroup.
        if (wave == 0) {
            tdm_load_2d(ldsK, kp + (size_t)l0 * D_, 16, 64);
            tdm_load_2d(ldsV, vp + (size_t)l0 * D_, 16, 64);
            __builtin_amdgcn_s_wait_tensorcnt(0);
        }
#else
#pragma unroll
        for (int i = 0; i < 2; ++i) {
            int idx = (tid + i * 256) * 8;
            *(float4*)(&Ks[idx]) = *(const float4*)(kp + (size_t)l0 * D_ + idx);
            *(float4*)(&Vs[idx]) = *(const float4*)(vp + (size_t)l0 * D_ + idx);
        }
#endif
        __syncthreads();

#pragma unroll
        for (int kc = 0; kc < 2; ++kc) {
            // A = K^T: A[m=d][k=l] = Ks[l][d]  -> transposed gather.
            v16h a = load_frag_colK(Ks, kc * 32, D_, tm, lane);
#pragma unroll
            for (int t = 0; t < 2; ++t) {
                v16h b = load_frag_colK(Vs, kc * 32, D_, tn0 + t * 16, lane);
                acc[t] = wmma_f16(a, b, acc[t]);
            }
        }
        __syncthreads();
    }

    float* ap = attn + (size_t)nh * D_ * D_;
    int mhi = (lane >> 4) ? 8 : 0, ncol = lane & 15;
#pragma unroll
    for (int t = 0; t < 2; ++t)
#pragma unroll
        for (int r = 0; r < 8; ++r)
            ap[(size_t)(tm + mhi + r) * D_ + tn0 + t * 16 + ncol] = acc[t][r];
}

// ---------------------------------------------------------------------------
// Kernel 4: out = l2norm(0.5 v + (1/pi) q@attn) + depthwise_conv9(v)
// One block per (n,h, 128-l tile); wave handles 16 l-rows (4 wmma N-tiles).
// Writes yh in [N, L, C] f16 for the projection GEMM.
// ---------------------------------------------------------------------------
__global__ __launch_bounds__(256)
void out_kernel(const _Float16* __restrict__ qh, const float* __restrict__ v32,
                const float* __restrict__ attn, const float* __restrict__ w_dconv,
                _Float16* __restrict__ yh) {
    __shared__ __align__(16) _Float16 Ah[64 * 64];    // attn as f16 [d][e]
    __shared__ __align__(16) float    Vs[136 * 64];   // v rows [l0-4, l0+132)

    int nh = blockIdx.x;
    int n  = nh >> 3, h = nh & 7;
    int l0 = blockIdx.y * 128;
    int tid = threadIdx.x, lane = tid & 31, wave = tid >> 5;

    // Stage attn f32 -> f16, packed pairs.
    const float* ap = attn + (size_t)nh * D_ * D_;
    for (int i = tid; i < 2048; i += 256) {
        const float2 av = *(const float2*)(ap + i * 2);
        Pack2 pk; pk.h[0] = (_Float16)av.x; pk.h[1] = (_Float16)av.y;
        ((unsigned int*)Ah)[i] = pk.u;
    }

    // Stage v rows with +-4 halo (zero padded), float2 moves.
    const float* vp = v32 + (size_t)nh * L_ * D_;
#pragma unroll
    for (int i = 0; i < 17; ++i) {
        int p = tid + i * 256;                        // 4352 pairs = 136*32
        int r = p >> 5, d2 = (p & 31) << 1;
        int gl = l0 - 4 + r;
        float2 vv = make_float2(0.0f, 0.0f);
        if (gl >= 0 && gl < L_) vv = *(const float2*)(vp + (size_t)gl * D_ + d2);
        *(float2*)(&Vs[r * 64 + d2]) = vv;
    }

    float wd[KS_];
#pragma unroll
    for (int t = 0; t < KS_; ++t) wd[t] = w_dconv[h * KS_ + t];
    __syncthreads();

    int lrow = l0 + wave * 16;
    const _Float16* qp = qh + ((size_t)nh * L_ + lrow) * D_;

    v8f acc[4] = {};
#pragma unroll
    for (int kc = 0; kc < 2; ++kc) {
        v16h a = load_frag_rowK(qp, 0, D_, kc * 32, lane);   // q rows, contiguous K
#pragma unroll
        for (int t = 0; t < 4; ++t) {
            v16h b = load_frag_colK(Ah, kc * 32, D_, t * 16, lane);
            acc[t] = wmma_f16(a, b, acc[t]);
        }
    }

    int mhi = (lane >> 4) ? 8 : 0, ncol = lane & 15;
    float ssq[8];
#pragma unroll
    for (int r = 0; r < 8; ++r) ssq[r] = 0.0f;

#pragma unroll
    for (int t = 0; t < 4; ++t) {
#pragma unroll
        for (int r = 0; r < 8; ++r) {
            int lr = wave * 16 + mhi + r;
            int d  = t * 16 + ncol;
            float tmp = 0.5f * Vs[(lr + 4) * 64 + d] + INV_PI_ * acc[t][r];
            acc[t][r] = tmp;
            ssq[r] += tmp * tmp;
        }
    }
#pragma unroll
    for (int r = 0; r < 8; ++r) {
#pragma unroll
        for (int m = 8; m >= 1; m >>= 1) ssq[r] += __shfl_xor(ssq[r], m, 32);
        ssq[r] = 1.0f / sqrtf(ssq[r]);
    }

#pragma unroll
    for (int t = 0; t < 4; ++t) {
#pragma unroll
        for (int r = 0; r < 8; ++r) {
            int lr = wave * 16 + mhi + r;
            int gl = l0 + lr;
            int d  = t * 16 + ncol;
            float dc = 0.0f;
#pragma unroll
            for (int u = 0; u < KS_; ++u)
                dc += wd[u] * Vs[(lr + u) * 64 + d];
            float outv = acc[t][r] * ssq[r] + dc;
            yh[((size_t)(n * L_ + gl)) * C_ + h * D_ + d] = (_Float16)outv;
        }
    }
}

// ---------------------------------------------------------------------------
// Kernel 5: out = yh @ w_proj + b_proj  (f16 A, f32 B->f16, f32 out).
// ---------------------------------------------------------------------------
__global__ __launch_bounds__(256)
void proj_gemm_kernel(const _Float16* __restrict__ yh, const float* __restrict__ w,
                      const float* __restrict__ bias, float* __restrict__ out) {
    __shared__ __align__(16) _Float16 As[128 * 32];
    __shared__ __align__(16) _Float16 Bs[64 * 32];   // transposed [N][K]

    int tid  = threadIdx.x;
    int lane = tid & 31, wave = tid >> 5;
    int bm = blockIdx.x * 128;
    int bn = blockIdx.y * 64;
    int wm = (wave & 3) * 32;
    int wn = (wave >> 2) * 32;

    v8f acc[2][2] = {};

    for (int kb = 0; kb < C_; kb += 32) {
        // Stage A: half -> half as dword copies.
        unsigned int* Asu = (unsigned int*)As;
#pragma unroll
        for (int i = 0; i < 8; ++i) {
            int uidx = tid + i * 256;
            int r = uidx >> 4, c2 = uidx & 15;
            Asu[uidx] = *(const unsigned int*)(yh + (size_t)(bm + r) * C_ + kb + c2 * 2);
        }
        // Stage B transposed, packed K pairs.
#pragma unroll
        for (int i = 0; i < 4; ++i) {
            int p = tid + i * 256;
            int n = p & 63, c2 = (p >> 6) << 1;
            float b0 = w[(size_t)(kb + c2) * C_ + bn + n];
            float b1 = w[(size_t)(kb + c2 + 1) * C_ + bn + n];
            Pack2 pk; pk.h[0] = (_Float16)b0; pk.h[1] = (_Float16)b1;
            ((unsigned int*)Bs)[n * 16 + (c2 >> 1)] = pk.u;
        }
        __syncthreads();

        v16h af[2], bf[2];
#pragma unroll
        for (int i = 0; i < 2; ++i) af[i] = load_frag_rowK(As, wm + i * 16, 32, 0, lane);
#pragma unroll
        for (int j = 0; j < 2; ++j) bf[j] = load_frag_rowK(Bs, wn + j * 16, 32, 0, lane);
#pragma unroll
        for (int i = 0; i < 2; ++i)
#pragma unroll
            for (int j = 0; j < 2; ++j)
                acc[i][j] = wmma_f16(af[i], bf[j], acc[i][j]);
        __syncthreads();
    }

    int mhi = (lane >> 4) ? 8 : 0, ncol = lane & 15;
#pragma unroll
    for (int i = 0; i < 2; ++i) {
#pragma unroll
        for (int j = 0; j < 2; ++j) {
#pragma unroll
            for (int r = 0; r < 8; ++r) {
                int m = bm + wm + i * 16 + mhi + r;
                int c = bn + wn + j * 16 + ncol;
                out[(size_t)m * C_ + c] = acc[i][j][r] + bias[c];
            }
        }
    }
}

// ---------------------------------------------------------------------------
// Launch
// ---------------------------------------------------------------------------
extern "C" void kernel_launch(void* const* d_in, const int* in_sizes, int n_in,
                              void* d_out, int out_size, void* d_ws, size_t ws_size,
                              hipStream_t stream) {
    const float* x       = (const float*)d_in[0];
    const float* w_qkv   = (const float*)d_in[1];
    const float* b_qkv   = (const float*)d_in[2];
    const float* w_dconv = (const float*)d_in[3];
    const float* w_proj  = (const float*)d_in[4];
    const float* b_proj  = (const float*)d_in[5];
    float* out = (float*)d_out;

    const size_t E = (size_t)NB_ * H_ * L_ * D_;   // 8,388,608 (= N*L*C)

    char* ws = (char*)d_ws;
    float*    q32  = (float*)ws;     ws += E * 4;
    float*    k32  = (float*)ws;     ws += E * 4;
    float*    v32  = (float*)ws;     ws += E * 4;
    _Float16* qh   = (_Float16*)ws;  ws += E * 2;
    _Float16* kh   = (_Float16*)ws;  ws += E * 2;
    _Float16* vh   = (_Float16*)ws;  ws += E * 2;
    float*    attn = (float*)ws;     ws += (size_t)NB_ * H_ * D_ * D_ * 4;
    _Float16* yh   = (_Float16*)ws;  ws += E * 2;

    dim3 blk(256);

    qkv_gemm_kernel<<<dim3((NB_ * L_) / 128, (3 * C_) / 64), blk, 0, stream>>>(
        x, w_qkv, b_qkv, q32, k32, v32);

    norm_cast_kernel<<<dim3((NB_ * H_ * L_) / 8), blk, 0, stream>>>(
        q32, k32, v32, qh, kh, vh);

    attn_kernel<<<dim3(NB_ * H_), blk, 0, stream>>>(kh, vh, attn);

    out_kernel<<<dim3(NB_ * H_, L_ / 128), blk, 0, stream>>>(
        qh, v32, attn, w_dconv, yh);

    proj_gemm_kernel<<<dim3((NB_ * L_) / 128, C_ / 64), blk, 0, stream>>>(
        yh, w_proj, b_proj, out);
}
